// Correlation_50869592654596
// MI455X (gfx1250) — matrix operand order repaired and checked by reading, not compiled, not run on hardware
//
#include <hip/hip_runtime.h>

// Problem constants (from reference): B=8, C=256, H=64, W=96, 21x21 even displacements.
#define B_   8
#define C_   256
#define H_   64
#define W_   96
#define HW_  (H_*W_)
#define ND_  21
#define NOUT_ (ND_*ND_)

// LDS layouts (ushort bf16 storage). Row stride C_+2 = 258 ushorts = 129 dwords
// -> consecutive parity-columns land on consecutive banks (129 mod 64 == 1).
#define IN1_STRIDE 258
#define IN1_PAR    (48*IN1_STRIDE)          // 48 same-parity columns of image1
#define IN2_Q      80                       // parity columns i0-10 .. i0+37 max => 80
#define IN2_STRIDE 258
#define IN2_PAR    (IN2_Q*IN2_STRIDE)
#define BAND_STRIDE 97                      // padded 96 to de-conflict banded scatter

typedef __attribute__((ext_vector_type(16))) __bf16 bf16x16;
typedef __attribute__((ext_vector_type(8)))  float  f32x8;

union AFrag { unsigned int u[8]; bf16x16 v; };

__device__ __forceinline__ unsigned short f32_to_bf16(float x) {
  unsigned int u = __float_as_uint(x);
  u += 0x7FFFu + ((u >> 16) & 1u);          // round-to-nearest-even
  return (unsigned short)(u >> 16);
}

__device__ __forceinline__ unsigned int pack_bf16x2(float lo, float hi) {
  return (unsigned int)f32_to_bf16(lo) | ((unsigned int)f32_to_bf16(hi) << 16);
}

__global__ __launch_bounds__(192) void corr441_wmma_bf16(
    const float* __restrict__ in1, const float* __restrict__ in2,
    float* __restrict__ out)
{
  __shared__ __attribute__((aligned(16))) unsigned short s1[2*IN1_PAR];
  __shared__ __attribute__((aligned(16))) unsigned short s2[2*IN2_PAR];
  __shared__ __attribute__((aligned(16))) float sband[ND_*BAND_STRIDE];

  const int h    = blockIdx.x;
  const int b    = blockIdx.y;
  const int tid  = threadIdx.x;
  const int lane = tid & 31;
  const int wid  = tid >> 5;        // 0..5
  const int par  = wid & 1;         // output-column parity owned by this wave
  const int mt   = wid >> 1;        // M tile 0..2  (16 same-parity pixels each)
  const int i0   = mt * 16;

  // ---- zero s2 halo columns once (qoff 0..9 and 58..79, both parities) ----
  for (int idx = tid; idx < 2*32*(C_/2); idx += 192) {   // u32 granularity
    int p  = idx / (32*(C_/2));
    int r  = idx - p*(32*(C_/2));
    int qi = r / (C_/2);
    int c2 = r - qi*(C_/2);
    int qoff = (qi < 10) ? qi : (qi + 48);
    *(unsigned int*)&s2[p*IN2_PAR + qoff*IN2_STRIDE + 2*c2] = 0u;
  }

  // ---- image1 row (b,:,h,:) -> bf16 LDS, parity-split (loaded once) ----
  {
    const float* src = in1 + ((size_t)b*C_*H_ + (size_t)h)*W_;
    for (int idx = tid; idx < (C_/4)*W_; idx += 192) {
      int c4 = idx / W_;
      int w  = idx - c4*W_;
      const float* g = src + (size_t)(4*c4)*HW_ + w;
      float v0 = g[0], v1 = g[HW_], v2 = g[2*HW_], v3 = g[3*HW_];
      unsigned short* d = &s1[(w & 1)*IN1_PAR + (w >> 1)*IN1_STRIDE + 4*c4];
      *(unsigned int*)(d + 0) = pack_bf16x2(v0, v1);
      *(unsigned int*)(d + 2) = pack_bf16x2(v2, v3);
    }
  }
  __syncthreads();

  // ---- preload this wave's A fragments: 16 rows x K=256 (8 WMMA K-steps) ----
  // A layout (contiguous, matches B): lanes 0-15 M=lane K{0-15}; lanes 16-31 same M K{16-31}
  AFrag afr[8];
  {
    const int mrow = lane & 15;
    const int klo  = (lane & 16) ? 16 : 0;
    const unsigned short* abase = &s1[par*IN1_PAR + (i0 + mrow)*IN1_STRIDE];
#pragma unroll
    for (int s = 0; s < 8; ++s) {
#pragma unroll
      for (int r = 0; r < 8; ++r) {
        int k = 32*s + klo + 2*r;
        afr[s].u[r] = *(const unsigned int*)(abase + k);
      }
    }
  }

  const f32x8 zacc = {0.f,0.f,0.f,0.f,0.f,0.f,0.f,0.f};
  const int ncol    = lane & 15;             // B/N column, C/D column
  const int mrow_hi = (lane & 16) ? 8 : 0;   // C/D row offset for upper lanes
  const int kb      = (lane & 16) ? 16 : 0;  // B: lanes 16-31 hold K=16-31
  const float* in2b = in2 + (size_t)b*C_*H_*W_;

  for (int dyi = 0; dyi < ND_; ++dyi) {
    __syncthreads();                          // previous iter fully consumed s2/sband
    const int h2 = h + 2*dyi - 20;
    if (h2 >= 0 && h2 < H_) {
      // image2 row (b,:,h2,:) -> bf16 LDS real columns (qoff 10..57),
      // 4 channels per item: packed bf16x2 dword stores (merge to ds_store_2addr_b32)
      const float* src = in2b + (size_t)h2*W_;
      for (int idx = tid; idx < (C_/4)*W_; idx += 192) {
        int c4 = idx / W_;
        int w  = idx - c4*W_;
        const float* g = src + (size_t)(4*c4)*HW_ + w;
        float v0 = g[0], v1 = g[HW_], v2 = g[2*HW_], v3 = g[3*HW_];
        unsigned short* d = &s2[(w & 1)*IN2_PAR + ((w >> 1) + 10)*IN2_STRIDE + 4*c4];
        *(unsigned int*)(d + 0) = pack_bf16x2(v0, v1);
        *(unsigned int*)(d + 2) = pack_bf16x2(v2, v3);
      }
      __syncthreads();

      // prefetch next dy's row into cache while the matrix cores work
      {
        const int h2n = h2 + 2;
        if (h2n < H_) {
          const float* pn = in2b + (size_t)h2n*W_;
#pragma unroll
          for (int i = 0; i < 4; ++i)   // 192 thr * 4 * 128B = 96KB = full row
            __builtin_prefetch(pn + (size_t)(tid + i*192)*32, 0, 1);
        }
      }

      // 3 N-tiles (band cover) x 8 K-steps on the matrix cores
      f32x8 acc0 = zacc, acc1 = zacc, acc2 = zacc;
#pragma unroll
      for (int s = 0; s < 8; ++s) {
        AFrag bf0, bf1, bf2;
        const unsigned short* bbase =
            &s2[par*IN2_PAR + (i0 + ncol)*IN2_STRIDE + 32*s + kb];  // qoff = i0+16j+n
#pragma unroll
        for (int r = 0; r < 8; ++r) bf0.u[r] = *(const unsigned int*)(bbase + 2*r);
#pragma unroll
        for (int r = 0; r < 8; ++r) bf1.u[r] = *(const unsigned int*)(bbase + 16*IN2_STRIDE + 2*r);
#pragma unroll
        for (int r = 0; r < 8; ++r) bf2.u[r] = *(const unsigned int*)(bbase + 32*IN2_STRIDE + 2*r);
        acc0 = __builtin_amdgcn_wmma_f32_16x16x32_bf16(false, afr[s].v, false, bf0.v,
                                                       (short)0, acc0, false, false);
        acc1 = __builtin_amdgcn_wmma_f32_16x16x32_bf16(false, afr[s].v, false, bf1.v,
                                                       (short)0, acc1, false, false);
        acc2 = __builtin_amdgcn_wmma_f32_16x16x32_bf16(false, afr[s].v, false, bf2.v,
                                                       (short)0, acc2, false, false);
      }

      // scatter the 21 useful diagonals (dx index = 16j + n - m) into the band buffer
#pragma unroll
      for (int r = 0; r < 8; ++r) {
        int m  = r + mrow_hi;
        int w1 = ((i0 + m) << 1) + par;
        int d0 = ncol - m;
        if ((unsigned)d0 < 21u) sband[d0*BAND_STRIDE + w1] = acc0[r] * (1.0f/256.0f);
        int d1 = d0 + 16;
        if ((unsigned)d1 < 21u) sband[d1*BAND_STRIDE + w1] = acc1[r] * (1.0f/256.0f);
        int d2 = d0 + 32;
        if ((unsigned)d2 < 21u) sband[d2*BAND_STRIDE + w1] = acc2[r] * (1.0f/256.0f);
      }
      __syncthreads();

      // coalesced store of this dy's [21 x 96] slab
      float* dst = out + (((size_t)b*NOUT_ + (size_t)dyi*ND_)*H_ + h)*W_;
      for (int idx = tid; idx < ND_*W_; idx += 192) {
        int jd = idx / W_;
        int w  = idx - jd*W_;
        dst[(size_t)jd*HW_ + w] = sband[jd*BAND_STRIDE + w];
      }
    } else {
      // h+dy entirely inside zero padding -> output slab is zero
      float* dst = out + (((size_t)b*NOUT_ + (size_t)dyi*ND_)*H_ + h)*W_;
      for (int idx = tid; idx < ND_*W_; idx += 192) {
        int jd = idx / W_;
        int w  = idx - jd*W_;
        dst[(size_t)jd*HW_ + w] = 0.0f;
      }
    }
  }
}

extern "C" void kernel_launch(void* const* d_in, const int* in_sizes, int n_in,
                              void* d_out, int out_size, void* d_ws, size_t ws_size,
                              hipStream_t stream) {
  (void)in_sizes; (void)n_in; (void)d_ws; (void)ws_size; (void)out_size;
  const float* in1 = (const float*)d_in[0];
  const float* in2 = (const float*)d_in[1];
  float* out = (float*)d_out;
  dim3 grid(H_, B_);
  dim3 block(192);
  hipLaunchKernelGGL(corr441_wmma_bf16, grid, block, 0, stream, in1, in2, out);
}
